// CompactCausalAttention_6966436954701
// MI455X (gfx1250) — matrix operand order repaired and checked by reading, not compiled, not run on hardware
//
#include <hip/hip_runtime.h>
#include <hip/hip_bf16.h>

// ---------------------------------------------------------------------------
// CDNA5 (gfx1250) causal attention forward.
//   Kernel 1: QKV projection GEMMs, fp32 -> bf16 WMMA -> bf16 (V stored transposed).
//   Kernel 2: flash attention, bf16 WMMA + fp32 online softmax; Q tile staged to
//             LDS via the Tensor Data Mover (tensor_load_to_lds + s_wait_tensorcnt)
//             when the toolchain exposes the builtin.
// ---------------------------------------------------------------------------

typedef __attribute__((ext_vector_type(16))) __bf16    v16bf;
typedef __attribute__((ext_vector_type(8)))  float     v8f;
typedef __attribute__((ext_vector_type(4)))  unsigned  u32x4;
typedef __attribute__((ext_vector_type(8)))  int       i32x8;
typedef __attribute__((ext_vector_type(4)))  int       i32x4;

union Frag { u32x4 q[2]; v16bf v; };

__device__ __forceinline__ unsigned short f2bf(float f) {
    union { float f; unsigned u; } c; c.f = f;
    unsigned u = c.u + 0x7FFFu + ((c.u >> 16) & 1u);   // round-to-nearest-even
    return (unsigned short)(u >> 16);
}

// Packed fp32x2 -> bf16x2 (lo, hi) — lowers to v_cvt_pk_bf16_f32 where available.
__device__ __forceinline__ unsigned f2bf2(float lo, float hi) {
    union { __hip_bfloat162 h; unsigned u; } c;
    c.h = __float22bfloat162_rn(make_float2(lo, hi));
    return c.u;
}

__device__ __forceinline__ v8f wmma_bf16(const Frag& a, const Frag& b, v8f c) {
    return __builtin_amdgcn_wmma_f32_16x16x32_bf16(
        false, a.v, false, b.v, (short)0, c, false, false);
}

#define BT   4
#define TT   2048
#define DD   1024
#define MTOT (BT * TT)   // 8192

#if __has_builtin(__builtin_amdgcn_tensor_load_to_lds) && \
    __has_builtin(__builtin_amdgcn_s_wait_tensorcnt)
#define HAVE_TDM 1
#if defined(__has_include)
#if __has_include(<hip/amd_detail/amd_gfx1250_TDM.h>)
#define TDM_SIX_ARG 1
#endif
#endif
#endif

// Qs LDS layout: row stride 1040 halves (2080B = 130*16B), with an 8-half (16B)
// gap after half 512 in each row — matches TDM pad_interval=256dw/pad_amount=4dw
// and gives 16B-aligned fragment chunks with reduced bank clustering.
#define QROW 1040
__device__ __forceinline__ const unsigned short*
q_elem(const unsigned short* Qs, int row, int h) {
    return Qs + row * QROW + h + ((h >= 512) ? 8 : 0);
}
__device__ __forceinline__ unsigned short*
q_elem_w(unsigned short* Qs, int row, int h) {
    return Qs + row * QROW + h + ((h >= 512) ? 8 : 0);
}

// ---------------------------------------------------------------------------
// Kernel 1: C[8192,1024] = X[8192,1024] @ W[1024,1024]; z selects Wq/Wk/Wv.
// Block tile 128x64, 8 waves, wave tile 32x32, K-step 32.
// z==0 -> Q bf16 row-major, z==1 -> K bf16 row-major, z==2 -> V^T bf16 [b][d][t].
// ---------------------------------------------------------------------------
__global__ __launch_bounds__(256)
void qkv_gemm(const float* __restrict__ X,
              const float* __restrict__ Wq,
              const float* __restrict__ Wk,
              const float* __restrict__ Wv,
              unsigned short* __restrict__ Qb,
              unsigned short* __restrict__ Kb,
              unsigned short* __restrict__ Vtb)
{
    const int z  = blockIdx.z;
    const float* W = (z == 0) ? Wq : (z == 1) ? Wk : Wv;

    const int m0   = blockIdx.y * 128;
    const int n0   = blockIdx.x * 64;
    const int tid  = threadIdx.x;
    const int lane = tid & 31;
    const int w    = tid >> 5;      // 0..7
    const int waveM = w >> 1;       // 0..3 (rows)
    const int waveN = w & 1;        // 0..1 (cols)

    __shared__ unsigned short As[128][40];   // [m][k], 80B rows (16B aligned)
    __shared__ unsigned short Bs[64][40];    // [n][k] (transposed W tile)

    v8f acc[2][2];
    #pragma unroll
    for (int i = 0; i < 2; i++)
        #pragma unroll
        for (int j = 0; j < 2; j++)
            #pragma unroll
            for (int e = 0; e < 8; e++) acc[i][j][e] = 0.0f;

    const int arow  = tid >> 1;          // 0..127
    const int ahalf = (tid & 1) * 16;    // 0 or 16
    const int bk    = tid >> 3;          // 0..31
    const int bg    = (tid & 7) * 8;     // 0..56

    const int r   = lane & 15;
    const int hi  = (lane >= 16);
    const int akb = hi ? 16 : 0;         // A frag k sub-offset
    const int bks = hi ? 16 : 0;         // B frag k sub-offset

    for (int k0 = 0; k0 < DD; k0 += 32) {
        __syncthreads();
        // ---- stage A tile (128x32 fp32 -> bf16, packed conversions) ----
        {
            const float* xs = X + (size_t)(m0 + arow) * DD + k0 + ahalf;
            union { unsigned d[8]; u32x4 q[2]; } t;
            #pragma unroll
            for (int j = 0; j < 8; j++) t.d[j] = f2bf2(xs[2 * j], xs[2 * j + 1]);
            *(u32x4*)&As[arow][ahalf]     = t.q[0];
            *(u32x4*)&As[arow][ahalf + 8] = t.q[1];
        }
        // ---- stage B tile transposed (32x64 fp32 -> bf16 [n][k]) ----
        {
            const float* wsrc = W + (size_t)(k0 + bk) * DD + n0 + bg;
            #pragma unroll
            for (int j = 0; j < 8; j++) Bs[bg + j][bk] = f2bf(wsrc[j]);
        }
        __syncthreads();

        #pragma unroll
        for (int mt = 0; mt < 2; mt++) {
            Frag a;
            const unsigned short* ap = &As[waveM * 32 + mt * 16 + r][akb];
            a.q[0] = *(const u32x4*)ap;
            a.q[1] = *(const u32x4*)(ap + 8);
            #pragma unroll
            for (int nt = 0; nt < 2; nt++) {
                Frag bf_;
                const unsigned short* bp = &Bs[waveN * 32 + nt * 16 + r][bks];
                bf_.q[0] = *(const u32x4*)bp;
                bf_.q[1] = *(const u32x4*)(bp + 8);
                acc[mt][nt] = wmma_bf16(a, bf_, acc[mt][nt]);
            }
        }
    }

    // ---- store epilogue (z branch hoisted; no per-element branching) ----
    if (z < 2) {
        unsigned short* C = (z == 0) ? Qb : Kb;
        #pragma unroll
        for (int mt = 0; mt < 2; mt++) {
            #pragma unroll
            for (int nt = 0; nt < 2; nt++) {
                const int row0 = m0 + waveM * 32 + mt * 16 + hi * 8;
                const int col  = n0 + waveN * 32 + nt * 16 + r;
                unsigned short* base = C + (size_t)row0 * DD + col;
                #pragma unroll
                for (int j = 0; j < 8; j++)
                    base[(size_t)j * DD] = f2bf(acc[mt][nt][j]);
            }
        }
    } else {
        const int bb = m0 >> 11;             // whole block is one batch (128 | 2048)
        const int t0 = m0 & 2047;
        #pragma unroll
        for (int mt = 0; mt < 2; mt++) {
            #pragma unroll
            for (int nt = 0; nt < 2; nt++) {
                const int col  = n0 + waveN * 32 + nt * 16 + r;              // d
                const int trow = t0 + waveM * 32 + mt * 16 + hi * 8;         // t
                unsigned* base =
                    (unsigned*)(Vtb + ((size_t)(bb * DD + col)) * TT + trow); // 16B-aligned
                #pragma unroll
                for (int j = 0; j < 4; j++)                                  // packed pairs
                    base[j] = f2bf2(acc[mt][nt][2 * j], acc[mt][nt][2 * j + 1]);
            }
        }
    }
}

// ---------------------------------------------------------------------------
// Kernel 2: flash attention. Block = 16 query rows, 256 threads (8 waves).
// kv tiles of 128; wave w owns S cols [w*16, w*16+16) and O cols [w*128, w*128+128).
// ---------------------------------------------------------------------------
__global__ __launch_bounds__(256)
void attn(const unsigned short* __restrict__ Qb,
          const unsigned short* __restrict__ Kb,
          const unsigned short* __restrict__ Vtb,
          float* __restrict__ Out)
{
    const int blk = blockIdx.x;          // 0..511
    const int b   = blk >> 7;            // batch
    const int q0  = (blk & 127) * 16;    // in-batch first query row

    const int tid  = threadIdx.x;
    const int lane = tid & 31;
    const int w    = tid >> 5;

    __shared__ unsigned short Qs[16 * QROW];   // staged Q (bf16), TDM-padded layout
    __shared__ float          Sbuf[16][132];   // fp32 scores
    __shared__ unsigned short Pbuf[16][136];   // bf16 probabilities
    __shared__ float          red[16][16];
    __shared__ float          mprev[16], alphaA[16], lA[16];

    // ---- stage Q block into LDS ----
#if defined(HAVE_TDM)
    if (w == 0) {
        // Tensor DMA: 16x1024 bf16 tile of Q -> LDS (padded), one descriptor.
        const unsigned lds_addr = (unsigned)(size_t)&Qs[0];
        const unsigned long long ga =
            (unsigned long long)(size_t)(Qb + ((size_t)(b * TT + q0)) * DD);
        u32x4 g0;
        g0[0] = 1u;                                      // count=1 (valid user D#)
        g0[1] = lds_addr;                                // lds_addr [63:32]
        g0[2] = (unsigned)(ga & 0xFFFFFFFFu);            // global_addr lo
        g0[3] = (unsigned)((ga >> 32) & 0x1FFFFFFu)      // global_addr hi (57-bit)
                | (2u << 30);                            // type=2 ("image")
        i32x8 g1;
        g1[0] = (1 << 16)        // data_size = 2B
              | (1 << 20)        // pad_enable
              | (7 << 22)        // pad_interval = 256 dwords
              | (3 << 25);       // pad_amount  = 4 dwords (16B)
        g1[1] = (int)(1024u << 16);   // tensor_dim0 = 1024 (lo16 @ bits 63:48)
        g1[2] = (int)(8192u << 16);   // tensor_dim1 = 8192 (lo16 @ bits 95:80)
        g1[3] = (int)(1024u << 16);   // tile_dim0   = 1024 (@ bits 127:112)
        g1[4] = 16;                   // tile_dim1   = 16   (@ bits 143:128)
        g1[5] = 1024;                 // tensor_dim0_stride = 1024 (lo32)
        g1[6] = 0;
        g1[7] = 0;
        i32x4 gz; gz[0] = 0; gz[1] = 0; gz[2] = 0; gz[3] = 0;
#if defined(TDM_SIX_ARG)
        i32x8 z8; z8[0]=0; z8[1]=0; z8[2]=0; z8[3]=0; z8[4]=0; z8[5]=0; z8[6]=0; z8[7]=0;
        __builtin_amdgcn_tensor_load_to_lds(g0, g1, gz, gz, z8, 0);
#else
        __builtin_amdgcn_tensor_load_to_lds(g0, g1, gz, gz, 0);
#endif
        __builtin_amdgcn_s_wait_tensorcnt(0);
    }
#else
    {
        const int row = tid >> 4;    // 0..15
        const int seg = tid & 15;    // 64 halves each
        const unsigned short* src = Qb + ((size_t)(b * TT + q0 + row)) * DD + seg * 64;
        #pragma unroll
        for (int j = 0; j < 8; j++) {
            const int h = seg * 64 + j * 8;
            *(u32x4*)q_elem_w(Qs, row, h) = *(const u32x4*)(src + j * 8);
        }
    }
#endif
    if (tid < 16) { mprev[tid] = -1e30f; lA[tid] = 0.0f; }
    __syncthreads();

    v8f o[8];
    #pragma unroll
    for (int nt = 0; nt < 8; nt++)
        #pragma unroll
        for (int e = 0; e < 8; e++) o[nt][e] = 0.0f;

    const int r   = lane & 15;
    const int hi  = (lane >= 16);
    const int akb = hi ? 16 : 0;
    const int bks = hi ? 16 : 0;
    const float scale = 0.022097086912079608f;   // 1/sqrt(2048) (faithful to ref)

    const int ntiles = (q0 + 16 + 127) / 128;
    for (int kt = 0; kt < ntiles; kt++) {
        const int kv0 = kt * 128;
        __syncthreads();   // guard Sbuf/Pbuf reuse across iterations

        // ---- S = Q K^T (wave's 16 kv columns), accumulate over D in 32-chunks ----
        v8f s;
        #pragma unroll
        for (int e = 0; e < 8; e++) s[e] = 0.0f;
        const int kvcol = kv0 + w * 16 + r;
        const unsigned short* krow = Kb + ((size_t)(b * TT + kvcol)) * DD;
        if (kt + 1 < ntiles) {  // prefetch next kv tile's K row (global_prefetch_b8)
            __builtin_prefetch(Kb + ((size_t)(b * TT + kvcol + 128)) * DD, 0, 1);
        }
        #pragma unroll 4
        for (int c = 0; c < 32; c++) {
            Frag a, bf_;
            a.q[0]   = *(const u32x4*)q_elem(Qs, r, c * 32 + akb);
            a.q[1]   = *(const u32x4*)q_elem(Qs, r, c * 32 + akb + 8);
            const unsigned short* bp = krow + c * 32 + bks;
            bf_.q[0] = *(const u32x4*)bp;
            bf_.q[1] = *(const u32x4*)(bp + 8);
            s = wmma_bf16(a, bf_, s);
        }
        // ---- store scaled + causal-masked scores ----
        #pragma unroll
        for (int j = 0; j < 8; j++) {
            const int row  = j + hi * 8;
            const int colg = kv0 + w * 16 + r;
            Sbuf[row][w * 16 + r] = (colg <= q0 + row) ? s[j] * scale : -1e30f;
        }
        __syncthreads();

        // ---- online softmax: row max ----
        {
            const int row = tid & 15;
            const int seg = tid >> 4;   // 8 cols each
            float mx = -1e30f;
            #pragma unroll
            for (int j = 0; j < 8; j++) mx = fmaxf(mx, Sbuf[row][seg * 8 + j]);
            red[row][seg] = mx;
        }
        __syncthreads();
        if (tid < 16) {
            float mt_ = red[tid][0];
            #pragma unroll
            for (int j = 1; j < 16; j++) mt_ = fmaxf(mt_, red[tid][j]);
            const float mnew = fmaxf(mprev[tid], mt_);
            alphaA[tid] = __expf(mprev[tid] - mnew);
            mprev[tid]  = mnew;
        }
        __syncthreads();
        // ---- P = exp(S - m), row sums (packed bf16 stores) ----
        {
            const int row = tid & 15;
            const int seg = tid >> 4;
            const float mnew = mprev[row];
            float ls = 0.0f;
            unsigned* pdst = (unsigned*)&Pbuf[row][seg * 8];
            #pragma unroll
            for (int j = 0; j < 4; j++) {
                const float p0 = __expf(Sbuf[row][seg * 8 + 2 * j]     - mnew);
                const float p1 = __expf(Sbuf[row][seg * 8 + 2 * j + 1] - mnew);
                ls += p0 + p1;
                pdst[j] = f2bf2(p0, p1);
            }
            red[row][seg] = ls;
        }
        __syncthreads();
        if (tid < 16) {
            float ssum = 0.0f;
            #pragma unroll
            for (int j = 0; j < 16; j++) ssum += red[tid][j];
            lA[tid] = alphaA[tid] * lA[tid] + ssum;
        }

        // ---- rescale O by alpha[row] ----
        float alj[8];
        #pragma unroll
        for (int j = 0; j < 8; j++) alj[j] = alphaA[j + hi * 8];
        #pragma unroll
        for (int nt = 0; nt < 8; nt++)
            #pragma unroll
            for (int j = 0; j < 8; j++) o[nt][j] *= alj[j];

        // ---- O += P @ V  (B-fragments straight from global V^T, L2-resident) ----
        #pragma unroll
        for (int c = 0; c < 4; c++) {
            Frag a;
            const unsigned short* ap = &Pbuf[r][c * 32 + akb];
            a.q[0] = *(const u32x4*)ap;
            a.q[1] = *(const u32x4*)(ap + 8);
            #pragma unroll
            for (int nt = 0; nt < 8; nt++) {
                Frag bf_;
                const int d = w * 128 + nt * 16 + r;
                const unsigned short* vp =
                    Vtb + ((size_t)(b * DD + d)) * TT + kv0 + c * 32 + bks;
                bf_.q[0] = *(const u32x4*)vp;
                bf_.q[1] = *(const u32x4*)(vp + 8);
                o[nt] = wmma_bf16(a, bf_, o[nt]);
            }
        }
    }
    __syncthreads();

    // ---- normalize and write fp32 output ----
    float rin[8];
    #pragma unroll
    for (int j = 0; j < 8; j++) rin[j] = 1.0f / lA[j + hi * 8];
    #pragma unroll
    for (int nt = 0; nt < 8; nt++) {
        #pragma unroll
        for (int j = 0; j < 8; j++) {
            const int row = q0 + j + hi * 8;
            const int d   = w * 128 + nt * 16 + r;
            Out[((size_t)(b * TT + row)) * DD + d] = o[nt][j] * rin[j];
        }
    }
}

// ---------------------------------------------------------------------------
extern "C" void kernel_launch(void* const* d_in, const int* in_sizes, int n_in,
                              void* d_out, int out_size, void* d_ws, size_t ws_size,
                              hipStream_t stream) {
    (void)in_sizes; (void)n_in; (void)out_size; (void)ws_size;
    const float* X  = (const float*)d_in[0];
    const float* Wq = (const float*)d_in[1];
    const float* Wk = (const float*)d_in[2];
    const float* Wv = (const float*)d_in[3];

    const size_t stride = (size_t)MTOT * DD;           // elements (bf16)
    unsigned short* qb  = (unsigned short*)d_ws;       // Q  bf16 [8192][1024]
    unsigned short* kb  = qb + stride;                 // K  bf16 [8192][1024]
    unsigned short* vtb = kb + stride;                 // V^T bf16 [4][1024][2048]

    dim3 g1(DD / 64, MTOT / 128, 3);
    qkv_gemm<<<g1, 256, 0, stream>>>(X, Wq, Wk, Wv, qb, kb, vtb);

    attn<<<dim3((BT * TT) / 16), 256, 0, stream>>>(qb, kb, vtb, (float*)d_out);
}
